// AlexNet_6571299962917
// MI455X (gfx1250) — matrix operand order repaired and checked by reading, not compiled, not run on hardware
//
#include <hip/hip_runtime.h>
#include <cmath>

// ---------------------------------------------------------------------------
// Deformable-conv pipeline for MI455X (gfx1250, wave32).
//   K1: conv1(1->3,3x3,pad1) + ReLU + maxpool(3x3,s2)  -> hp [32,3,255,255] (ws)
//   K2: fused offset-conv(3->27 via WMMA f16 GEMM, K=27 padded to 32)
//       + sigmoid mask + zero-padded bilinear gather + DCN contraction.
// The 225MB "om" tensor is never materialized: each 27x16 tile lives in
// registers/LDS, so DRAM traffic is ~110MB for the whole problem (~5us at
// 23.3 TB/s); the pipeline is latency-bound, so this round removes the
// exec-mask churn around the WMMAs (branchless gathers, hoisted biases,
// LDS-staged DCN weights).
// ---------------------------------------------------------------------------

typedef __attribute__((ext_vector_type(16))) _Float16 v16h;
typedef __attribute__((ext_vector_type(8)))  float    v8f;

#define HO 255
#define WO 255
#define HI 512
#define WI 512
#define NB 32
#define PLANE (HO * WO)

// ------------------------------ Kernel 1 -----------------------------------
__global__ __launch_bounds__(256) void dcn_conv_relu_pool(
    const float* __restrict__ x,      // [32,1,512,512]
    const float* __restrict__ w,      // [3,1,3,3]
    const float* __restrict__ bias,   // [3]
    float* __restrict__ hp)           // [32,3,255,255]
{
    int idx = blockIdx.x * blockDim.x + threadIdx.x;
    const int total = NB * HO * WO;
    if (idx >= total) return;
    int xo = idx % WO;
    int t  = idx / WO;
    int yo = t % HO;
    int b  = t / HO;

    // 5x5 input patch covering the 3x3 pool window of 3x3 convs.
    // Branchless: clamp address, select zero where out of bounds.
    const float* xb = x + b * HI * WI;
    float p[5][5];
    int r0 = 2 * yo - 1, c0 = 2 * xo - 1;
#pragma unroll
    for (int i = 0; i < 5; ++i) {
        int r  = r0 + i;
        int rc = min(max(r, 0), HI - 1);
#pragma unroll
        for (int j = 0; j < 5; ++j) {
            int c  = c0 + j;
            int cc = min(max(c, 0), WI - 1);
            float v = xb[rc * WI + cc];
            bool ok = (r >= 0) && (r < HI) && (c >= 0) && (c < WI);
            p[i][j] = ok ? v : 0.f;
        }
    }
#pragma unroll
    for (int ch = 0; ch < 3; ++ch) {
        float wv[9];
#pragma unroll
        for (int q = 0; q < 9; ++q) wv[q] = w[ch * 9 + q];
        float bv = bias[ch];
        float mx = -1e30f;
#pragma unroll
        for (int wy = 0; wy < 3; ++wy) {
#pragma unroll
            for (int wx = 0; wx < 3; ++wx) {
                float s = bv;
#pragma unroll
                for (int ky = 0; ky < 3; ++ky)
#pragma unroll
                    for (int kx = 0; kx < 3; ++kx)
                        s = fmaf(wv[ky * 3 + kx], p[wy + ky][wx + kx], s);
                s = fmaxf(s, 0.f);
                mx = fmaxf(mx, s);
            }
        }
        hp[((b * 3 + ch) * HO + yo) * WO + xo] = mx;
    }
}

// ------------------------------ Kernel 2 -----------------------------------
// One wave handles a tile of 16 consecutive pixels in a row.
// Phase 1: offset-conv as WMMA GEMM: om[27x16] = W[27x27] * im2col[27x16]
//          (K padded to 32, M split into two 16-row WMMA tiles).
// Phase 2: deformable sampling + DCN contraction.
#define TPR 16                       // tiles per row: ceil(255/16)
#define WAVES_PER_BLOCK 8

__global__ __launch_bounds__(256) void dcn_fused_offset_sample(
    const float* __restrict__ hp,     // [32,3,255,255]
    const float* __restrict__ off_w,  // [27,3,3,3] -> A[27][27]
    const float* __restrict__ off_b,  // [27]
    const float* __restrict__ dcn_w,  // [3,3,3,3]  -> [o][c][k]
    const float* __restrict__ dcn_b,  // [3]
    float* __restrict__ out)          // [32,3,255,255]
{
    __shared__ float sh[WAVES_PER_BLOCK][32][16];   // per-wave om tile (16 KB)
    __shared__ float sdcnw[81];                     // dcn_w [o][c][k]
    __shared__ float sdcnb[3];

    // stage DCN weights once per block
    if (threadIdx.x < 81) sdcnw[threadIdx.x] = dcn_w[threadIdx.x];
    if (threadIdx.x < 3)  sdcnb[threadIdx.x] = dcn_b[threadIdx.x];

    const int lane  = threadIdx.x & 31;
    const int wave  = threadIdx.x >> 5;
    const int mrow  = lane & 15;      // A: row of M-tile held by this lane
    const int khalf = lane >> 4;      // lane-half selects K sub-range
    const int mbase = 8 * khalf;      // C: rows written by this lane

    // ---- Build A (weight) fragments once per wave. ISA 16-bit 16x32 A layout:
    // slot j -> VGPR j>>1, K = (vgpr>=4?16:0) + 2*(vgpr&3) + (j&1) + 8*khalf.
    v16h a0{}, a1{};
#pragma unroll
    for (int j = 0; j < 16; ++j) {
        int vg  = j >> 1;
        int Kk  = ((vg >= 4) ? 16 : 0) + 2 * (vg & 3) + (j & 1) + 8 * khalf;
        float v0 = (Kk < 27) ? off_w[mrow * 27 + Kk] : 0.f;
        int m1 = 16 + mrow;
        float v1 = (Kk < 27 && m1 < 27) ? off_w[m1 * 27 + Kk] : 0.f;
        a0[j] = (_Float16)v0;
        a1[j] = (_Float16)v1;
    }

    // ---- Hoist the offset-conv bias values this lane will add (rows
    // mbase..mbase+7 of each C tile); loaded once, reused every tile.
    float bq0[8], bq1[8];
#pragma unroll
    for (int r = 0; r < 8; ++r) {
        int m  = mbase + r;           // 0..15, always valid
        int m2 = 16 + mbase + r;      // 16..31, valid < 27
        bq0[r] = off_b[m];
        bq1[r] = (m2 < 27) ? off_b[m2] : 0.f;
    }

    const int tiles_total = NB * HO * TPR;                       // 130560
    const int waves_total = gridDim.x * WAVES_PER_BLOCK;
    const int wave_gid    = blockIdx.x * WAVES_PER_BLOCK + wave;
    const int iters       = (tiles_total + waves_total - 1) / waves_total;

    __syncthreads();                  // sdcnw ready

    for (int it = 0; it < iters; ++it) {
        int tile = wave_gid + it * waves_total;
        bool active = (tile < tiles_total);                      // wave-uniform

        int b = 0, y = 0, x0 = 0;
        if (active) {
            b = tile / (HO * TPR);
            int rem = tile % (HO * TPR);
            y  = rem / TPR;
            x0 = (rem % TPR) * 16;
        }
        const float* __restrict__ hpb = hp + b * 3 * PLANE;

        // ---- Phase 1: im2col B fragment + 2x WMMA -> om tile in LDS --------
        if (active) {
            const int n    = lane & 15;
            const int xpix = x0 + n;
            // B 32x16 layout (matches A): slot j -> K = j + 8*khalf for j<8,
            // 16 + 8*khalf + (j-8) for j>=8; column = lane&15.
            // Branchless gather: clamp address, cndmask-select zero.
            v16h bm{};
#pragma unroll
            for (int j = 0; j < 16; ++j) {
                int Kk = (j < 8) ? (j + 8 * khalf) : (16 + 8 * khalf + (j - 8));
                int c  = Kk / 9;
                int tt = Kk % 9;
                int yy = y    - 1 + tt / 3;
                int xx = xpix - 1 + tt % 3;
                bool ok = (Kk < 27) && (xpix < WO) &&
                          (yy >= 0) && (yy < HO) && (xx >= 0) && (xx < WO);
                int cc  = min(c, 2);
                int yyc = min(max(yy, 0), HO - 1);
                int xxc = min(max(xx, 0), WO - 1);
                float v = hpb[(cc * HO + yyc) * WO + xxc];
                bm[j] = (_Float16)(ok ? v : 0.f);
            }

            v8f cz = {};
            v8f acc0 = __builtin_amdgcn_wmma_f32_16x16x32_f16(
                false, a0, false, bm, (short)0, cz, false, false);
            v8f acc1 = __builtin_amdgcn_wmma_f32_16x16x32_f16(
                false, a1, false, bm, (short)0, cz, false, false);

            // C layout: lane col n = lane&15; VGPR r -> row r + 8*khalf.
#pragma unroll
            for (int r = 0; r < 8; ++r) {
                sh[wave][mbase + r][n]      = acc0[r] + bq0[r];
                sh[wave][16 + mbase + r][n] = acc1[r] + bq1[r];
            }
        }
        __syncthreads();

        // ---- Phase 2: deformable bilinear sampling + DCN contraction -------
        float acc[3] = {0.f, 0.f, 0.f};
        if (active) {
            const int p    = lane & 15;
            const int xp   = x0 + p;
            const int kbeg = (lane < 16) ? 0 : 5;
            const int kend = (lane < 16) ? 5 : 9;

            for (int k = kbeg; k < kend; ++k) {
                float dy  = sh[wave][k][p];
                float dx  = sh[wave][9 + k][p];
                float msk = 1.f / (1.f + __expf(-sh[wave][18 + k][p]));
                float py  = (float)y  - 1.f + (float)(k / 3) + dy;
                float px  = (float)xp - 1.f + (float)(k % 3) + dx;
                float y0f = floorf(py), x0f = floorf(px);
                float ly  = py - y0f,   lx  = px - x0f;

                float cw[4] = { (1.f - ly) * (1.f - lx), (1.f - ly) * lx,
                                ly * (1.f - lx),         ly * lx };
                float cy[4] = { y0f, y0f, y0f + 1.f, y0f + 1.f };
                float cx[4] = { x0f, x0f + 1.f, x0f, x0f + 1.f };

                float sval[3] = {0.f, 0.f, 0.f};
#pragma unroll
                for (int q = 0; q < 4; ++q) {
                    // Branchless corner: clamp index, zero the weight if OOB.
                    bool ok = (cy[q] >= 0.f) && (cy[q] <= (float)(HO - 1)) &&
                              (cx[q] >= 0.f) && (cx[q] <= (float)(WO - 1));
                    float wq = ok ? cw[q] : 0.f;
                    int yi = (int)fminf(fmaxf(cy[q], 0.f), (float)(HO - 1));
                    int xi = (int)fminf(fmaxf(cx[q], 0.f), (float)(WO - 1));
                    int base = yi * WO + xi;
#pragma unroll
                    for (int c = 0; c < 3; ++c)
                        sval[c] = fmaf(wq, hpb[c * PLANE + base], sval[c]);
                }
#pragma unroll
                for (int c = 0; c < 3; ++c) {
                    float sm = sval[c] * msk;
#pragma unroll
                    for (int o = 0; o < 3; ++o)
                        acc[o] = fmaf(sdcnw[o * 27 + c * 9 + k], sm, acc[o]);
                }
            }

            // combine tap-halves: lanes l and l^16 hold the same pixel
#pragma unroll
            for (int o = 0; o < 3; ++o)
                acc[o] += __shfl_xor(acc[o], 16, 32);

            if (lane < 16 && xp < WO) {
#pragma unroll
                for (int o = 0; o < 3; ++o)
                    out[((b * 3 + o) * HO + y) * WO + xp] = acc[o] + sdcnb[o];
            }
        }
        __syncthreads();
    }
}

// ------------------------------ Launch -------------------------------------
extern "C" void kernel_launch(void* const* d_in, const int* in_sizes, int n_in,
                              void* d_out, int out_size, void* d_ws, size_t ws_size,
                              hipStream_t stream) {
    const float* x       = (const float*)d_in[0];
    const float* conv1_w = (const float*)d_in[1];
    const float* conv1_b = (const float*)d_in[2];
    const float* off_w   = (const float*)d_in[3];
    const float* off_b   = (const float*)d_in[4];
    const float* dcn_w   = (const float*)d_in[5];
    const float* dcn_b   = (const float*)d_in[6];
    float* out = (float*)d_out;
    float* hp  = (float*)d_ws;            // needs 32*3*255*255*4 = ~25 MB

    // K1: one thread per pooled pixel
    int total1 = NB * HO * WO;
    int blocks1 = (total1 + 255) / 256;
    hipLaunchKernelGGL(dcn_conv_relu_pool, dim3(blocks1), dim3(256), 0, stream,
                       x, conv1_w, conv1_b, hp);

    // K2: 130560 tiles; 1632 blocks * 8 waves = 13056 waves -> exactly 10 iters
    hipLaunchKernelGGL(dcn_fused_offset_sample, dim3(1632), dim3(256), 0, stream,
                       hp, off_w, off_b, dcn_w, dcn_b, out);
}
